// LVPNet_42339787604107
// MI455X (gfx1250) — matrix operand order repaired and checked by reading, not compiled, not run on hardware
//
#include <hip/hip_runtime.h>
#include <hip/hip_bf16.h>

// ---------------------------------------------------------------------------
// LVPNet forward for MI455X (gfx1250, wave32, WMMA bf16 16x16x32).
// Refinement 3x3 convs run as implicit-GEMM on the matrix pipes, bf16 in /
// f32 accumulate. v2: 2Mx2N wave micro-tiles (fragment reuse: 8 ds_load_b128
// per 4 WMMA), bank-conflict-free padded LDS strides, 103KB LDS tile using
// CDNA5's 320KB WGP LDS. Tiny down-path / latent stages are scalar kernels.
// ---------------------------------------------------------------------------

typedef __bf16 v16bf __attribute__((ext_vector_type(16)));
typedef float  v8f   __attribute__((ext_vector_type(8)));

#define H_IMG 192
#define W_IMG 192
#define NB 4
#define CREF 64
#define STEPQ 0.01f

#define NCOL 66          // 64-px tile + 2 halo columns
#define CIS  72          // padded ci stride (halves): 36 dwords, bank-friendly
#define WROW 584         // padded weight row stride (halves): 292 dwords
#define SMEM_HALVES (64 * WROW + 3 * NCOL * CIS)   // 37376 + 14256 = 51632
#define SMEM_BYTES  (SMEM_HALVES * 2)              // 103264 B

__device__ __forceinline__ unsigned short f2bf(float f) {
    unsigned u = __builtin_bit_cast(unsigned, f);
    unsigned r = u + 0x7FFFu + ((u >> 16) & 1u);   // round-to-nearest-even
    return (unsigned short)(r >> 16);
}

union Frag { uint4 q[2]; v16bf v; };

// A 16-bit layout: elems 0..7 -> K = 8g..8g+7 ; elems 8..15 -> K = 16+8g..
__device__ __forceinline__ void load_frag_a(Frag& f, const unsigned short* p, int g) {
    f.q[0] = *reinterpret_cast<const uint4*>(p + 8 * g);
    f.q[1] = *reinterpret_cast<const uint4*>(p + 16 + 8 * g);
}
// B 16-bit layout: elems 0..7 -> K = 8g..8g+7 ; elems 8..15 -> K = 16+8g..
__device__ __forceinline__ void load_frag_b(Frag& f, const unsigned short* p, int g) {
    f.q[0] = *reinterpret_cast<const uint4*>(p + 8 * g);
    f.q[1] = *reinterpret_cast<const uint4*>(p + 16 + 8 * g);
}

#define WMMA_BF16(a, b, c) \
    __builtin_amdgcn_wmma_f32_16x16x32_bf16(false, (a), false, (b), (short)0, (c), false, false)

// ---------------- down path (tiny, scalar) ----------------

__global__ void k_down1(const float* __restrict__ x, const float* __restrict__ w,
                        float* __restrict__ out) {
    int idx = blockIdx.x * blockDim.x + threadIdx.x;      // 4*16*48*48
    if (idx >= 4 * 16 * 48 * 48) return;
    int ow = idx % 48; int t = idx / 48;
    int oh = t % 48; t /= 48;
    int co = t % 16; int n = t / 16;
    const float* xin = x + (size_t)n * H_IMG * W_IMG;
    const float* wp = w + co * 16;
    float acc = 0.f;
    int ih = oh * 4, iw = ow * 4;
#pragma unroll
    for (int kh = 0; kh < 4; ++kh)
#pragma unroll
        for (int kw = 0; kw < 4; ++kw)
            acc += xin[(ih + kh) * W_IMG + iw + kw] * wp[kh * 4 + kw];
    out[idx] = fmaxf(acc, 0.f);
}

__global__ void k_down2(const float* __restrict__ h1, const float* __restrict__ w,
                        float* __restrict__ out) {
    int idx = blockIdx.x * blockDim.x + threadIdx.x;      // 4*256*12*12
    if (idx >= 4 * 256 * 12 * 12) return;
    int ow = idx % 12; int t = idx / 12;
    int oh = t % 12; t /= 12;
    int co = t % 256; int n = t / 256;
    float acc = 0.f;
    for (int ci = 0; ci < 16; ++ci) {
        const float* ip = h1 + ((size_t)(n * 16 + ci) * 48 + oh * 4) * 48 + ow * 4;
        const float* wp = w + ((size_t)co * 16 + ci) * 16;
#pragma unroll
        for (int kh = 0; kh < 4; ++kh)
#pragma unroll
            for (int kw = 0; kw < 4; ++kw)
                acc += ip[kh * 48 + kw] * wp[kh * 4 + kw];
    }
    out[idx] = fmaxf(acc, 0.f);
}

__global__ void k_down3(const float* __restrict__ h2, const float* __restrict__ w,
                        float* __restrict__ out) {
    int idx = blockIdx.x * blockDim.x + threadIdx.x;      // 4*1024*6*6
    if (idx >= 4 * 1024 * 6 * 6) return;
    int ow = idx % 6; int t = idx / 6;
    int oh = t % 6; t /= 6;
    int co = t % 1024; int n = t / 1024;
    float acc = 0.f;
    for (int ci = 0; ci < 256; ++ci) {
        const float* ip = h2 + ((size_t)(n * 256 + ci) * 12 + oh * 2) * 12 + ow * 2;
        const float* wp = w + ((size_t)co * 256 + ci) * 4;
        acc += ip[0] * wp[0] + ip[1] * wp[1] + ip[12] * wp[2] + ip[13] * wp[3];
    }
    out[idx] = fmaxf(acc, 0.f);
}

// ---------------- latent: d4 + quantize + compensation + up + depth2space ----

__global__ void k_latent(const float* __restrict__ h3, const float* __restrict__ d4w,
                         const float* __restrict__ q1w, const float* __restrict__ q1b,
                         const float* __restrict__ q2w, const float* __restrict__ q2b,
                         const float* __restrict__ q3w, const float* __restrict__ q3b,
                         const float* __restrict__ upw, const float* __restrict__ upb,
                         float* __restrict__ yb, float* __restrict__ out_comp,
                         float* __restrict__ out_x4) {
    __shared__ float hv[1024];
    __shared__ float lat[160];
    __shared__ float comp[160];
    int p = blockIdx.x;                    // 144 pixels: n*36 + h*6 + w
    int n = p / 36; int hw = p % 36; int h = hw / 6; int w = hw % 6;
    int co = threadIdx.x;

    for (int c = co; c < 1024; c += 256)
        hv[c] = h3[((size_t)(n * 1024 + c) * 6 + h) * 6 + w];
    __syncthreads();

    float xr = 0.f;
    if (co < 154) {
        const float* wr = d4w + (size_t)co * 1024;
        float acc = 0.f;
        for (int ci = 0; ci < 1024; ++ci) acc += wr[ci] * hv[ci];
        out_x4[((size_t)(n * 154 + co) * 6 + h) * 6 + w] = acc;
        xr = floorf(acc / STEPQ) * STEPQ;   // STE quantize-dequantize (forward)
        lat[co] = xr;
    }
    __syncthreads();

    for (int l = 0; l < 3; ++l) {
        const float* qw = (l == 0) ? q1w : (l == 1) ? q2w : q3w;
        const float* qb = (l == 0) ? q1b : (l == 1) ? q2b : q3b;
        float v = 0.f;
        if (co < 154) {
            float a = qb[co];
            const float* wr = qw + (size_t)co * 154;
            for (int ci = 0; ci < 154; ++ci) a += wr[ci] * lat[ci];
            v = fmaxf(a, 0.f) + lat[co];
        }
        __syncthreads();
        if (co < 154) lat[co] = v;
        __syncthreads();
    }

    if (co < 154) {
        float xc = lat[co] + xr;
        out_comp[((size_t)(n * 154 + co) * 6 + h) * 6 + w] = xc;
        comp[co] = xc;
    }
    __syncthreads();

    // 1x1 up conv (154 -> 1024) fused with depth_to_space(32)
    for (int c = co; c < 1024; c += 256) {
        const float* wr = upw + (size_t)c * 154;
        float a = upb[c];
        for (int ci = 0; ci < 154; ++ci) a += wr[ci] * comp[ci];
        int yy = h * 32 + (c >> 5);
        int yx = w * 32 + (c & 31);
        yb[((size_t)n * H_IMG + yy) * W_IMG + yx] = a;
    }
}

// ---------------- c1: 1->64 3x3 (K=9, too small for WMMA) -------------------

__global__ void k_c1(const float* __restrict__ y, const float* __restrict__ w,
                     const float* __restrict__ b, const float* __restrict__ ap,
                     float* __restrict__ out) {
    size_t idx = (size_t)blockIdx.x * blockDim.x + threadIdx.x;   // 4*64*192*192
    if (idx >= (size_t)NB * CREF * H_IMG * W_IMG) return;
    int ww = idx % W_IMG; size_t t = idx / W_IMG;
    int hh = t % H_IMG; t /= H_IMG;
    int co = t % CREF; int n = t / CREF;
    const float* yp = y + (size_t)n * H_IMG * W_IMG;
    const float* wp = w + co * 9;
    float acc = b[co];
#pragma unroll
    for (int kh = 0; kh < 3; ++kh) {
        int ih = hh + kh - 1;
        if ((unsigned)ih >= (unsigned)H_IMG) continue;
#pragma unroll
        for (int kw = 0; kw < 3; ++kw) {
            int iw = ww + kw - 1;
            if ((unsigned)iw >= (unsigned)W_IMG) continue;
            acc += yp[ih * W_IMG + iw] * wp[kh * 3 + kw];
        }
    }
    float al = ap[0];
    out[idx] = (acc >= 0.f) ? acc : al * acc;
}

// ---------------- weight convert+reorder: OIHW f32 -> [co][tap][ci] bf16 ----

__global__ void k_cvtw(const float* __restrict__ w, unsigned short* __restrict__ out,
                       int Cout) {
    int idx = blockIdx.x * blockDim.x + threadIdx.x;   // Cout*576
    if (idx >= Cout * 576) return;
    int ci = idx & 63;
    int tp = (idx >> 6) % 9;
    int co = idx / 576;
    out[idx] = f2bf(w[((size_t)co * 64 + ci) * 9 + tp]);
}

// ---------------- main WMMA implicit-GEMM 3x3 conv --------------------------
// Block: 128 thr (4 waves). Block tile: 64 Cout x 64 px, Cin=64, K = 9x64.
// Each wave owns a 2Mx2N micro-tile (32 co x 32 px): per K-step, 2 A-frags +
// 2 B-frags (8 ds_load_b128) feed 4 WMMAs -> 1KB LDS per WMMA (balanced).
// LDS 103KB (of 320KB/WGP); strides padded to kill bank conflicts.

__global__ __launch_bounds__(128)
void k_conv3x3(const float* __restrict__ inA, const float* __restrict__ inB,
               int bcastB, const unsigned short* __restrict__ wbf,
               const float* __restrict__ bias, const float* __restrict__ alpha_p,
               int use_prelu, float* __restrict__ out, int Cout) {
    extern __shared__ __align__(16) unsigned short smem[];
    unsigned short* s_w  = smem;                 // [64][WROW]
    unsigned short* s_in = smem + 64 * WROW;     // [3*NCOL][CIS]

    const int tid = threadIdx.x;
    const int w0  = blockIdx.x * 64;
    const int n   = blockIdx.y / H_IMG;
    const int h   = blockIdx.y % H_IMG;
    const int co0 = blockIdx.z * 64;

    // ---- stage input tile (fused residual add), padded ci stride ----
    for (int i = tid; i < 3 * NCOL * 64; i += 128) {
        int ci = i & 63;
        int cr = i >> 6;                 // row*NCOL + col
        int col = cr % NCOL;
        int row = cr / NCOL;
        int ih = h + row - 1;
        int iw = w0 + col - 1;
        float v = 0.f;
        if ((unsigned)ih < (unsigned)H_IMG && (unsigned)iw < (unsigned)W_IMG) {
            size_t gi = ((size_t)(n * CREF + ci) * H_IMG + ih) * W_IMG + iw;
            v = inA[gi];
            if (inB) {
                size_t gb = bcastB ? (((size_t)n * H_IMG + ih) * W_IMG + iw) : gi;
                v += inB[gb];
            }
        }
        s_in[cr * CIS + ci] = f2bf(v);
    }
    // ---- stage weights (bf16 [co][tap][ci], 72 uint4 per co row) ----
    for (int i = tid; i < 64 * 72; i += 128) {
        int co = i / 72;
        int j  = (i % 72) * 8;
        *reinterpret_cast<uint4*>(s_w + (size_t)co * WROW + j) =
            *reinterpret_cast<const uint4*>(wbf + (size_t)(co0 + co) * 576 + j);
    }
    __syncthreads();

    const int lane = tid & 31;
    const int wave = tid >> 5;       // 4 waves
    const int wm = wave & 1;         // co half: wm*32
    const int wn = wave >> 1;        // px half: wn*32
    const int g  = lane >> 4;
    const int lm = lane & 15;

    v8f a00 = {0.f, 0.f, 0.f, 0.f, 0.f, 0.f, 0.f, 0.f};
    v8f a01 = {0.f, 0.f, 0.f, 0.f, 0.f, 0.f, 0.f, 0.f};
    v8f a10 = {0.f, 0.f, 0.f, 0.f, 0.f, 0.f, 0.f, 0.f};
    v8f a11 = {0.f, 0.f, 0.f, 0.f, 0.f, 0.f, 0.f, 0.f};

    const unsigned short* ar0 = s_w + (size_t)(wm * 32 + lm) * WROW;  // M-tile 0
    const unsigned short* ar1 = ar0 + 16 * WROW;                      // M-tile 1
    const int n0 = wn * 32;

    for (int t = 0; t < 9; ++t) {
        const int kh = t / 3, kw = t % 3;
        const unsigned short* bc0 = s_in + (size_t)(kh * NCOL + n0 + lm + kw) * CIS;
        const unsigned short* bc1 = bc0 + 16 * CIS;                   // N-tile 1
#pragma unroll
        for (int cb = 0; cb < 2; ++cb) {
            Frag A0, A1, B0, B1;
            load_frag_a(A0, ar0 + t * 64 + cb * 32, g);
            load_frag_a(A1, ar1 + t * 64 + cb * 32, g);
            load_frag_b(B0, bc0 + cb * 32, g);
            load_frag_b(B1, bc1 + cb * 32, g);
            a00 = WMMA_BF16(A0.v, B0.v, a00);
            a01 = WMMA_BF16(A0.v, B1.v, a01);
            a10 = WMMA_BF16(A1.v, B0.v, a10);
            a11 = WMMA_BF16(A1.v, B1.v, a11);
        }
    }

    // ---- epilogue: bias + (prelu) + store ----
    const float al = use_prelu ? alpha_p[0] : 0.f;
    auto store_tile = [&](const v8f& a, int mi, int ni) {
        const int ow = w0 + n0 + ni * 16 + lm;          // N = lane&15
#pragma unroll
        for (int r = 0; r < 8; ++r) {
            int co = co0 + wm * 32 + mi * 16 + 8 * g + r;   // M = r + 8*(lane>=16)
            float v = a[r] + bias[co];
            if (use_prelu) v = (v >= 0.f) ? v : al * v;
            out[((size_t)(n * Cout + co) * H_IMG + h) * W_IMG + ow] = v;
        }
    };
    store_tile(a00, 0, 0);
    store_tile(a01, 0, 1);
    store_tile(a10, 1, 0);
    store_tile(a11, 1, 1);
}

// ---------------------------------------------------------------------------

extern "C" void kernel_launch(void* const* d_in, const int* in_sizes, int n_in,
                              void* d_out, int out_size, void* d_ws, size_t ws_size,
                              hipStream_t stream) {
    const float* x    = (const float*)d_in[0];
    const float* d1w  = (const float*)d_in[2];
    const float* d2w  = (const float*)d_in[3];
    const float* d3w  = (const float*)d_in[4];
    const float* d4w  = (const float*)d_in[5];
    const float* q1w  = (const float*)d_in[6];  const float* q1b = (const float*)d_in[7];
    const float* q2w  = (const float*)d_in[8];  const float* q2b = (const float*)d_in[9];
    const float* q3w  = (const float*)d_in[10]; const float* q3b = (const float*)d_in[11];
    const float* upw  = (const float*)d_in[12]; const float* upb = (const float*)d_in[13];
    const float* c1w  = (const float*)d_in[14]; const float* c1b = (const float*)d_in[15];
    const float* c1a  = (const float*)d_in[16];
    const float* c2w  = (const float*)d_in[17]; const float* c2b = (const float*)d_in[18];
    const float* c2a  = (const float*)d_in[19];
    const float* c3w  = (const float*)d_in[20]; const float* c3b = (const float*)d_in[21];
    const float* c3a  = (const float*)d_in[22];
    const float* c4w  = (const float*)d_in[23]; const float* c4b = (const float*)d_in[24];
    const float* c4a  = (const float*)d_in[25];
    const float* c5w  = (const float*)d_in[26]; const float* c5b = (const float*)d_in[27];

    const size_t b5_elems  = (size_t)NB * 256 * H_IMG * W_IMG;   // 37,748,736
    const size_t lat_elems = (size_t)NB * 154 * 6 * 6;           // 22,176
    float* out_b5   = (float*)d_out;
    float* out_comp = out_b5 + b5_elems;
    float* out_x4   = out_comp + lat_elems;

    // workspace carve-up
    const size_t IMG  = (size_t)NB * H_IMG * W_IMG;              // 147,456
    const size_t FMAP = (size_t)NB * CREF * H_IMG * W_IMG;       // 9,437,184
    float* h1   = (float*)d_ws;
    float* h2   = h1 + IMG;
    float* h3   = h2 + IMG;
    float* yb   = h3 + IMG;
    float* bufA = yb + IMG;            // b1, later b4
    float* bufB = bufA + FMAP;         // b2
    float* bufC = bufB + FMAP;         // b3
    unsigned short* wbf2 = (unsigned short*)(bufC + FMAP);
    unsigned short* wbf3 = wbf2 + 64 * 576;
    unsigned short* wbf4 = wbf3 + 64 * 576;
    unsigned short* wbf5 = wbf4 + 64 * 576;

    // allow >64KB dynamic LDS (CDNA5 WGP has 320KB)
    static_assert(SMEM_BYTES == 103264, "LDS size");
    hipFuncSetAttribute(reinterpret_cast<const void*>(&k_conv3x3),
                        hipFuncAttributeMaxDynamicSharedMemorySize, SMEM_BYTES);

    // weight convert + reorder (bf16 [co][tap][ci])
    k_cvtw<<<(64 * 576 + 255) / 256, 256, 0, stream>>>(c2w, wbf2, 64);
    k_cvtw<<<(64 * 576 + 255) / 256, 256, 0, stream>>>(c3w, wbf3, 64);
    k_cvtw<<<(64 * 576 + 255) / 256, 256, 0, stream>>>(c4w, wbf4, 64);
    k_cvtw<<<(256 * 576 + 255) / 256, 256, 0, stream>>>(c5w, wbf5, 256);

    // down path
    k_down1<<<(4 * 16 * 48 * 48) / 256, 256, 0, stream>>>(x, d1w, h1);
    k_down2<<<(4 * 256 * 12 * 12) / 256, 256, 0, stream>>>(h1, d2w, h2);
    k_down3<<<(4 * 1024 * 6 * 6) / 256, 256, 0, stream>>>(h2, d3w, h3);

    // latent: d4 + quantize + 3x comp + up + depth_to_space -> yb, x_comp, x4
    k_latent<<<NB * 6 * 6, 256, 0, stream>>>(h3, d4w, q1w, q1b, q2w, q2b, q3w, q3b,
                                             upw, upb, yb, out_comp, out_x4);

    // refinement CNN
    k_c1<<<(unsigned)(FMAP / 256), 256, 0, stream>>>(yb, c1w, c1b, c1a, bufA); // b1

    dim3 blk(128);
    dim3 g64(3, NB * H_IMG, 1);   // 64 Cout
    dim3 g256(3, NB * H_IMG, 4);  // 256 Cout
    // c2: in = b1 + broadcast(y)
    k_conv3x3<<<g64, blk, SMEM_BYTES, stream>>>(bufA, yb, 1, wbf2, c2b, c2a, 1, bufB, 64);
    // c3: in = b2 + b1
    k_conv3x3<<<g64, blk, SMEM_BYTES, stream>>>(bufB, bufA, 0, wbf3, c3b, c3a, 1, bufC, 64);
    // c4: in = b3 + b2 -> writes b4 into b1's buffer (b1 dead)
    k_conv3x3<<<g64, blk, SMEM_BYTES, stream>>>(bufC, bufB, 0, wbf4, c4b, c4a, 1, bufA, 64);
    // c5: in = b4 + b3 -> final 256-channel output straight to d_out
    k_conv3x3<<<g256, blk, SMEM_BYTES, stream>>>(bufA, bufC, 0, wbf5, c5b, c1a, 0, out_b5, 256);
}
